// SiT_Model_1348619731172
// MI455X (gfx1250) — matrix operand order repaired and checked by reading, not compiled, not run on hardware
//
#include <hip/hip_runtime.h>
#include <hip/hip_bf16.h>
#include <math.h>

#define B_ 16
#define C1_ 3
#define C2_ 10
#define IMG_ 224
#define WIN_ 112
#define PP 16
#define DIM_ 512
#define HEADS_ 8
#define HD_ 64
#define INNER_ 512
#define MLP_ 1024
#define N1_ 197
#define N2_ 50
#define SCALE_ 0.125f

typedef __attribute__((ext_vector_type(16))) _Float16 v16h;
typedef __attribute__((ext_vector_type(8)))  _Float16 v8h;
typedef __attribute__((ext_vector_type(8)))  float    v8f;

// ---------------------------------------------------------------------------
// Generic batched WMMA GEMM:  C = act(scale * A·B(^T) + bias) + R
// A: M x K (row major, lda); B: K x N (ldb) or N x K if transB.
// 256 threads = 8 waves; block tile 64(M) x 128(N); wave = 16(M) x 64(N);
// K-step 32; f32 staged to f16 in LDS; B stored transposed (BsT[n][k]) so
// both fragment types are contiguous 16-byte ds_load_b128 pairs.
// Staging uses clamped addresses + multiply-by-mask so loads are
// unconditional (no exec-masked serialization, batched loadcnt waits).
// ---------------------------------------------------------------------------
__global__ __launch_bounds__(256)
void gemm_wmma(const float* __restrict__ A, const float* __restrict__ Bm,
               const float* __restrict__ bias, const float* __restrict__ R,
               float* __restrict__ C,
               int M, int N, int K,
               int lda, int ldb, int ldc, int ldr,
               long sAb, long sAh, long sBb, long sBh,
               long sCb, long sCh, long sRb, long sRh,
               int Hn, int transB, int act, float scale)
{
    __shared__ _Float16 As[64][40];     // [m][k]   (80B rows, 16B-aligned chunks)
    __shared__ _Float16 BsT[128][40];   // [n][k]

    const int z  = blockIdx.z;
    const int bb = z / Hn, hh = z % Hn;
    const float* Ap = A  + bb * sAb + hh * sAh;
    const float* Bp = Bm + bb * sBb + hh * sBh;
    float*       Cp = C  + bb * sCb + hh * sCh;
    const float* Rp = R ? (R + bb * sRb + hh * sRh) : (const float*)0;

    const int mBase = blockIdx.y * 64;
    const int nBase = blockIdx.x * 128;
    const int tid   = threadIdx.x;
    const int wave  = tid >> 5;
    const int lane  = tid & 31;
    const int m0    = (wave >> 1) * 16;   // wave M sub-tile (0..48)
    const int nb    = (wave & 1) * 64;    // wave N half (0/64)

    v8f acc[4] = {};

    // --- staging thread mappings (branchless clamped addressing) ---
    // A: 64 rows x 32 cols, each thread: 1 row, 8 contiguous cols
    const int   ar   = tid >> 2;
    const int   acb  = (tid & 3) << 3;
    const int   agr  = mBase + ar;
    const int   arc  = (agr < M) ? agr : (M - 1);
    const float amsk = (agr < M) ? 1.0f : 0.0f;
    // B (NN): 32 k-rows x 128 cols, each thread: 1 k-row, 16 contiguous cols
    const int   bkr  = tid >> 3;
    const int   bcb  = (tid & 7) << 4;
    // B (NT): 128 n-rows x 32 k, each thread: 1 n-row, 16 contiguous k
    const int   tr   = tid >> 1;
    const int   tkb  = (tid & 1) << 4;
    const int   tgn  = nBase + tr;
    const int   trc  = (tgn < N) ? tgn : (N - 1);
    const float tmsk = (tgn < N) ? 1.0f : 0.0f;

    for (int k0 = 0; k0 < K; k0 += 32) {
        // ---- stage A tile (f32 -> f16): unconditional clamped loads * mask ----
        {
            const float* s = Ap + (long)arc * lda;
            if (k0 + 32 < K) __builtin_prefetch(s + k0 + 32 + acb, 0, 1);
            v8h h;
            #pragma unroll
            for (int j = 0; j < 8; ++j) {
                int gc = k0 + acb + j;
                int cc = (gc < K) ? gc : (K - 1);
                float msk = (gc < K) ? amsk : 0.0f;
                h[j] = (_Float16)(s[cc] * msk);
            }
            *(v8h*)&As[ar][acb] = h;
        }
        // ---- stage B tile ----
        if (transB) {
            // Bp is N x K: BsT[n][k] = Bp[n][k] -> contiguous reads AND stores
            const float* s = Bp + (long)trc * ldb;
            if (k0 + 32 < K) __builtin_prefetch(s + k0 + 32 + tkb, 0, 1);
            v8h h0, h1;
            #pragma unroll
            for (int j = 0; j < 8; ++j) {
                int gc = k0 + tkb + j;
                int cc = (gc < K) ? gc : (K - 1);
                float msk = (gc < K) ? tmsk : 0.0f;
                h0[j] = (_Float16)(s[cc] * msk);
            }
            #pragma unroll
            for (int j = 0; j < 8; ++j) {
                int gc = k0 + tkb + 8 + j;
                int cc = (gc < K) ? gc : (K - 1);
                float msk = (gc < K) ? tmsk : 0.0f;
                h1[j] = (_Float16)(s[cc] * msk);
            }
            *(v8h*)&BsT[tr][tkb]     = h0;
            *(v8h*)&BsT[tr][tkb + 8] = h1;
        } else {
            // Bp is K x N: coalesced row read, transposed store into BsT
            int   gk   = k0 + bkr;
            int   krc  = (gk < K) ? gk : (K - 1);
            float kmsk = (gk < K) ? 1.0f : 0.0f;
            const float* s = Bp + (long)krc * ldb;
            #pragma unroll
            for (int j = 0; j < 16; ++j) {
                int gn = nBase + bcb + j;
                int nc = (gn < N) ? gn : (N - 1);
                float msk = (gn < N) ? kmsk : 0.0f;
                BsT[bcb + j][bkr] = (_Float16)(s[nc] * msk);
            }
        }
        __syncthreads();

        // ---- fragments (2x ds_load_b128 each) + 4 WMMA ----
        const int arow = m0 + (lane & 15);
        const int akb  = (lane < 16) ? 0 : 8;
        v8h a0 = *(const v8h*)&As[arow][akb];
        v8h a1 = *(const v8h*)&As[arow][akb + 16];
        v16h af = __builtin_shufflevector(a0, a1, 0,1,2,3,4,5,6,7,8,9,10,11,12,13,14,15);
        const int bcol = lane & 15;
        const int bkb  = (lane < 16) ? 0 : 8;
        #pragma unroll
        for (int st = 0; st < 4; ++st) {
            const _Float16* bp = &BsT[nb + st * 16 + bcol][0];
            v8h b0 = *(const v8h*)(bp + bkb);
            v8h b1 = *(const v8h*)(bp + bkb + 16);
            v16h bf = __builtin_shufflevector(b0, b1, 0,1,2,3,4,5,6,7,8,9,10,11,12,13,14,15);
            acc[st] = __builtin_amdgcn_wmma_f32_16x16x32_f16(false, af, false, bf, (short)0, acc[st], false, false);
        }
        __syncthreads();
    }

    // ---- epilogue: VGPR r holds row (lane<16 ? r : r+8), col = lane%16 ----
    const int ccol = lane & 15;
    const int rb   = (lane < 16) ? 0 : 8;
    #pragma unroll
    for (int st = 0; st < 4; ++st) {
        int gn = nBase + nb + st * 16 + ccol;
        if (gn >= N) continue;
        float bz = bias ? bias[gn] : 0.0f;
        #pragma unroll
        for (int r = 0; r < 8; ++r) {
            int gm = mBase + m0 + rb + r;
            if (gm >= M) continue;
            float v = acc[st][r] * scale + bz;
            if (act == 1) v = 0.5f * v * (1.0f + erff(v * 0.70710678f)); // exact GELU
            if (Rp) v += Rp[(long)gm * ldr + gn];
            Cp[(long)gm * ldc + gn] = v;
        }
    }
}

// ---------------------------------------------------------------------------
// Grouped conv2d (stride 1, 'same' output), y = conv(x,w)+bias
// ---------------------------------------------------------------------------
__global__ void conv2d_k(const float* __restrict__ X, const float* __restrict__ W,
                         const float* __restrict__ bias, float* __restrict__ Y,
                         int Bn, int Cin, int Cout, int H, int Wd,
                         int kh, int kw, int ph, int pw, int G)
{
    long i = blockIdx.x * (long)blockDim.x + threadIdx.x;
    long total = (long)Bn * Cout * H * Wd;
    if (i >= total) return;
    int x = i % Wd; long t = i / Wd;
    int y = t % H;  t /= H;
    int co = t % Cout; int b = t / Cout;
    int cpg = Cin / G;
    int opg = Cout / G;
    int g = co / opg;
    float acc = bias[co];
    for (int ci = 0; ci < cpg; ++ci) {
        int cin = g * cpg + ci;
        for (int u = 0; u < kh; ++u) {
            int yy = y + u - ph;
            if (yy < 0 || yy >= H) continue;
            for (int v = 0; v < kw; ++v) {
                int xx = x + v - pw;
                if (xx < 0 || xx >= Wd) continue;
                acc += X[(((long)b * Cin + cin) * H + yy) * Wd + xx]
                     * W[(((long)co * cpg + ci) * kh + u) * kw + v];
            }
        }
    }
    Y[i] = acc;
}

// per-channel batch-stat reduction (mean, biased var)
__global__ __launch_bounds__(256)
void bn_stats_k(const float* __restrict__ Y, float* mean, float* var,
                int Bn, int C, long HW)
{
    __shared__ float rs[256], rq[256];
    int c = blockIdx.x, tid = threadIdx.x;
    long n = (long)Bn * HW;
    float s = 0.f, q = 0.f;
    for (long i = tid; i < n; i += 256) {
        long b = i / HW, hw = i % HW;
        float v = Y[(b * C + c) * HW + hw];
        s += v; q += v * v;
    }
    rs[tid] = s; rq[tid] = q; __syncthreads();
    for (int st = 128; st > 0; st >>= 1) {
        if (tid < st) { rs[tid] += rs[tid + st]; rq[tid] += rq[tid + st]; }
        __syncthreads();
    }
    if (tid == 0) { float mu = rs[0] / n; mean[c] = mu; var[c] = rq[0] / n - mu * mu; }
}

// BN (eps 1e-3) + affine + mish
__global__ void bn_mish_k(const float* __restrict__ Y, const float* mean, const float* var,
                          const float* g, const float* be, float* __restrict__ O,
                          int C, long HW, long total)
{
    long i = blockIdx.x * (long)blockDim.x + threadIdx.x;
    if (i >= total) return;
    int c = (int)((i / HW) % C);
    float y = (Y[i] - mean[c]) * rsqrtf(var[c] + 1e-3f) * g[c] + be[c];
    float sp = (y > 20.f) ? y : log1pf(__expf(y));
    O[i] = y * tanhf(sp);
}

// b c (h p1)(w p2) -> (b, h*w, (p1*P+p2)*C + c)   (contiguous GEMM A matrix)
__global__ void patchify_k(const float* __restrict__ X, float* __restrict__ A,
                           int Bn, int C, int H, int Wd)
{
    long i = blockIdx.x * (long)blockDim.x + threadIdx.x;
    int h = H / PP, w = Wd / PP;
    long total = (long)Bn * h * w * PP * PP * C;
    if (i >= total) return;
    int c = i % C; long t = i / C;
    int p2 = t % PP; t /= PP;
    int p1 = t % PP; t /= PP;
    int tw = t % w;  t /= w;
    int th = t % h;  int b = t / h;
    A[i] = X[(((long)b * C + c) * H + th * PP + p1) * Wd + tw * PP + p2];
}

// X[b,0,:] = cls + pos[0];  X[b,1+t,:] = E[b*hw+t,:] + pos[1+t]
__global__ void assemble_k(const float* __restrict__ E, const float* __restrict__ cls,
                           const float* __restrict__ pos, float* __restrict__ X,
                           int Bn, int N)
{
    long i = blockIdx.x * (long)blockDim.x + threadIdx.x;
    long total = (long)Bn * N * DIM_;
    if (i >= total) return;
    int c = i % DIM_; long t = i / DIM_;
    int n = t % N;    int b = t / N;
    float v = (n == 0) ? cls[c] : E[((long)b * (N - 1) + (n - 1)) * DIM_ + c];
    X[i] = v + pos[(long)n * DIM_ + c];
}

// row layernorm (eps 1e-5)
__global__ __launch_bounds__(256)
void layernorm_k(const float* __restrict__ X, const float* g, const float* b,
                 float* __restrict__ Y, int C)
{
    __shared__ float red[256];
    long row = blockIdx.x;
    const float* p = X + row * (long)C;
    float* q = Y + row * (long)C;
    int tid = threadIdx.x;
    float s = 0.f;
    for (int c = tid; c < C; c += 256) s += p[c];
    red[tid] = s; __syncthreads();
    for (int st = 128; st > 0; st >>= 1) { if (tid < st) red[tid] += red[tid + st]; __syncthreads(); }
    float mu = red[0] / C; __syncthreads();
    float v = 0.f;
    for (int c = tid; c < C; c += 256) { float d = p[c] - mu; v += d * d; }
    red[tid] = v; __syncthreads();
    for (int st = 128; st > 0; st >>= 1) { if (tid < st) red[tid] += red[tid + st]; __syncthreads(); }
    float inv = rsqrtf(red[0] / C + 1e-5f);
    for (int c = tid; c < C; c += 256) q[c] = (p[c] - mu) * inv * g[c] + b[c];
}

// in-place row softmax
__global__ __launch_bounds__(256)
void softmax_rows(float* __restrict__ X, int cols)
{
    __shared__ float red[256];
    long row = blockIdx.x;
    float* p = X + row * (long)cols;
    int tid = threadIdx.x;
    float mx = -1e30f;
    for (int c = tid; c < cols; c += 256) mx = fmaxf(mx, p[c]);
    red[tid] = mx; __syncthreads();
    for (int s = 128; s > 0; s >>= 1) { if (tid < s) red[tid] = fmaxf(red[tid], red[tid + s]); __syncthreads(); }
    mx = red[0]; __syncthreads();
    float sum = 0.f;
    for (int c = tid; c < cols; c += 256) { float e = __expf(p[c] - mx); p[c] = e; sum += e; }
    red[tid] = sum; __syncthreads();
    for (int s = 128; s > 0; s >>= 1) { if (tid < s) red[tid] += red[tid + s]; __syncthreads(); }
    float inv = 1.f / red[0];
    for (int c = tid; c < cols; c += 256) p[c] *= inv;
}

// MS = s1 + s2  (cols 1536..2047 of the two qkv buffers)
__global__ void msadd_k(const float* __restrict__ Q1, const float* __restrict__ Q2,
                        float* __restrict__ MS, int N)
{
    long i = blockIdx.x * (long)blockDim.x + threadIdx.x;
    long total = (long)B_ * N * INNER_;
    if (i >= total) return;
    int c = i % INNER_; long t = i / INNER_;
    long off = t * (4L * INNER_) + 3L * INNER_ + c;
    MS[i] = Q1[off] + Q2[off];
}

// token-wise depthwise conv1d (k=5, pad 2) over DIM axis + sigmoid gate
__global__ void gate_k(const float* __restrict__ MS, const float* __restrict__ cw,
                       const float* __restrict__ cb, float* __restrict__ O, int N)
{
    long i = blockIdx.x * (long)blockDim.x + threadIdx.x;
    long total = (long)B_ * N * INNER_;
    if (i >= total) return;
    int c = i % INNER_; long t = i / INNER_;
    int n = (int)(t % N);
    const float* row = MS + t * (long)INNER_;
    float a = cb[n];
    #pragma unroll
    for (int u = 0; u < 5; ++u) {
        int cc = c + u - 2;
        if (cc >= 0 && cc < INNER_) a += row[cc] * cw[n * 5 + u];
    }
    O[i] = MS[i] / (1.f + __expf(-a));
}

// out[b,:] = Xa[b,0,:] + Xb[b,0,:]
__global__ void pool_k(const float* __restrict__ Xa, const float* __restrict__ Xb,
                       float* __restrict__ out, int N)
{
    int i = blockIdx.x * blockDim.x + threadIdx.x;
    if (i >= B_ * DIM_) return;
    int c = i % DIM_, b = i / DIM_;
    out[i] = Xa[(long)b * N * DIM_ + c] + Xb[(long)b * N * DIM_ + c];
}

__global__ void concat_k(const float* __restrict__ a, const float* __restrict__ b,
                         float* __restrict__ o)
{
    int i = blockIdx.x * blockDim.x + threadIdx.x;
    if (i >= B_ * 2 * DIM_) return;
    int c = i % (2 * DIM_), bb = i / (2 * DIM_);
    o[i] = (c < DIM_) ? a[bb * DIM_ + c] : b[bb * DIM_ + (c - DIM_)];
}

// ---------------------------------------------------------------------------
extern "C" void kernel_launch(void* const* d_in, const int* in_sizes, int n_in,
                              void* d_out, int out_size, void* d_ws, size_t ws_size,
                              hipStream_t stream)
{
    (void)in_sizes; (void)n_in; (void)out_size; (void)ws_size;

    auto in = [&](int i) { return (const float*)d_in[i]; };
    const float* img1 = in(0);
    const float* img2 = in(1);

    // ---- workspace layout (floats) ----
    float* Wb = (float*)d_ws;
    long off = 0;
    auto alloc = [&](long n) { float* p = Wb + off; off += n; return p; };
    const long XSZ = (long)B_ * N1_ * DIM_;            // 1,613,824 (sized for N1 everywhere)
    float* X11 = alloc(XSZ);
    float* X12 = alloc(XSZ);
    float* X21 = alloc(XSZ);
    float* X22 = alloc(XSZ);
    float* xn1 = alloc(XSZ);
    float* xn2 = alloc(XSZ);
    float* smallR = alloc(65536);
    float* mean = smallR;          // 32
    float* var  = smallR + 32;     // 32
    float* xspe = smallR + 64;     // 8192
    float* xspa = xspe + B_ * DIM_;
    float* xcat = xspa + B_ * DIM_;          // 16384
    float* hh   = xcat + B_ * 2 * DIM_;      // 16384 (head hidden)
    long scratch = off;
    // conv-phase region (dead once transformer starts)
    float* convT0 = Wb + scratch;
    float* convT1 = convT0 + 2408448;        // 16*3*224*224
    float* patchA = convT1 + 2408448;
    float* embedE = patchA + 2408448;        // <= 3136*512
    // transformer-phase region (reuses conv region)
    float* qkv1 = Wb + scratch;
    float* qkv2 = qkv1 + (long)B_ * N1_ * 4 * INNER_;
    float* d1   = qkv2 + (long)B_ * N1_ * 4 * INNER_;
    float* d2   = d1 + (long)B_ * HEADS_ * N1_ * N1_;
    float* d3   = d2 + (long)B_ * HEADS_ * N1_ * N1_;
    float* MS   = d3 + (long)B_ * HEADS_ * N1_ * N1_;
    float* MSg  = MS  + XSZ;
    float* t3a  = MSg + XSZ;
    float* t3b  = t3a + XSZ;
    float* o1m  = t3b + XSZ;
    float* o2m  = o1m + XSZ;
    float* mlph = o2m + XSZ;                 // B*N1*1024

    auto gemm = [&](const float* A, const float* Bm, const float* bias, const float* R,
                    float* C, int M, int N, int K,
                    int lda, int ldb, int ldc, int ldr,
                    long sAb, long sAh, long sBb, long sBh,
                    long sCb, long sCh, long sRb, long sRh,
                    int Hn, int Zb, int transB, int act, float scale) {
        dim3 grid((unsigned)((N + 127) / 128), (unsigned)((M + 63) / 64), (unsigned)Zb);
        gemm_wmma<<<grid, 256, 0, stream>>>(A, Bm, bias, R, C, M, N, K,
                                            lda, ldb, ldc, ldr,
                                            sAb, sAh, sBb, sBh, sCb, sCh, sRb, sRh,
                                            Hn, transB, act, scale);
    };
    auto blocks = [](long n) { return (unsigned)((n + 255) / 256); };

    // ---- conv stems + patch embed + token assembly ----
    auto convStack = [&](const float* img, int Cc, int Hh, int Ww,
                         int cb0, int kh, int kw, int ph, int pw,
                         int pei, int clsi, int posi, float* X, int N) {
        long HW = (long)Hh * Ww;
        long tot = (long)B_ * Cc * HW;
        conv2d_k<<<blocks(tot), 256, 0, stream>>>(img, in(cb0 + 0), in(cb0 + 1), convT0,
                                                  B_, Cc, Cc, Hh, Ww, kh, kw, ph, pw, Cc);
        bn_stats_k<<<Cc, 256, 0, stream>>>(convT0, mean, var, B_, Cc, HW);
        bn_mish_k<<<blocks(tot), 256, 0, stream>>>(convT0, mean, var, in(cb0 + 2), in(cb0 + 3),
                                                   convT1, Cc, HW, tot);
        conv2d_k<<<blocks(tot), 256, 0, stream>>>(convT1, in(cb0 + 4), in(cb0 + 5), convT0,
                                                  B_, Cc, Cc, Hh, Ww, 1, 1, 0, 0, 1);
        bn_stats_k<<<Cc, 256, 0, stream>>>(convT0, mean, var, B_, Cc, HW);
        bn_mish_k<<<blocks(tot), 256, 0, stream>>>(convT0, mean, var, in(cb0 + 6), in(cb0 + 7),
                                                   convT1, Cc, HW, tot);
        patchify_k<<<blocks(tot), 256, 0, stream>>>(convT1, patchA, B_, Cc, Hh, Ww);
        int hw = (Hh / PP) * (Ww / PP);
        int F  = PP * PP * Cc;
        gemm(patchA, in(pei), in(pei + 1), nullptr, embedE,
             B_ * hw, DIM_, F, F, DIM_, DIM_, 0,
             0, 0, 0, 0, 0, 0, 0, 0, 1, 1, 0, 0, 1.f);
        long xt = (long)B_ * N * DIM_;
        assemble_k<<<blocks(xt), 256, 0, stream>>>(embedE, in(clsi), in(posi), X, B_, N);
    };

    convStack(img1, C1_, IMG_, IMG_,  2, 3, 1, 1, 0, 34, 44, 42, X11, N1_);
    convStack(img1, C1_, IMG_, IMG_, 10, 1, 3, 0, 1, 36, 45, 42, X12, N1_);
    convStack(img2, C2_, WIN_, WIN_, 18, 1, 1, 0, 0, 38, 46, 43, X21, N2_);
    convStack(img2, C2_, WIN_, WIN_, 26, 3, 3, 1, 1, 40, 47, 43, X22, N2_);

    // ---- transformer layer ----
    auto runLayer = [&](float* X1, float* X2, int N, int pb) {
        const long MN = (long)B_ * N;
        const long Q4 = 4L * INNER_;
        // LN1 (shared params for both streams)
        layernorm_k<<<(unsigned)MN, 256, 0, stream>>>(X1, in(pb + 0), in(pb + 1), xn1, DIM_);
        layernorm_k<<<(unsigned)MN, 256, 0, stream>>>(X2, in(pb + 0), in(pb + 1), xn2, DIM_);
        // QKV(S) projections  (B*N x 512) x (512 x 2048)
        gemm(xn1, in(pb + 2), nullptr, nullptr, qkv1, (int)MN, 4 * INNER_, DIM_,
             DIM_, 4 * INNER_, 4 * INNER_, 0, 0, 0, 0, 0, 0, 0, 0, 0, 1, 1, 0, 0, 1.f);
        gemm(xn2, in(pb + 3), nullptr, nullptr, qkv2, (int)MN, 4 * INNER_, DIM_,
             DIM_, 4 * INNER_, 4 * INNER_, 0, 0, 0, 0, 0, 0, 0, 0, 0, 1, 1, 0, 0, 1.f);
        // scores d1 = q1·k1^T, d2 = q2·k2^T  (batched over b,h;  A·B^T with K=64)
        gemm(qkv1, qkv1 + INNER_, nullptr, nullptr, d1, N, N, HD_,
             (int)Q4, (int)Q4, N, 0,
             (long)N * Q4, HD_, (long)N * Q4, HD_,
             (long)HEADS_ * N * N, (long)N * N, 0, 0,
             HEADS_, B_ * HEADS_, 1, 0, SCALE_);
        gemm(qkv2, qkv2 + INNER_, nullptr, nullptr, d2, N, N, HD_,
             (int)Q4, (int)Q4, N, 0,
             (long)N * Q4, HD_, (long)N * Q4, HD_,
             (long)HEADS_ * N * N, (long)N * N, 0, 0,
             HEADS_, B_ * HEADS_, 1, 0, SCALE_);
        // MS = s1+s2 ; conv-gate ; d3 = MSg·MSg^T
        long mt = MN * INNER_;
        msadd_k<<<blocks(mt), 256, 0, stream>>>(qkv1, qkv2, MS, N);
        gate_k<<<blocks(mt), 256, 0, stream>>>(MS, in(pb + 8), in(pb + 9), MSg, N);
        gemm(MSg, MSg, nullptr, nullptr, d3, N, N, HD_,
             INNER_, INNER_, N, 0,
             (long)N * INNER_, HD_, (long)N * INNER_, HD_,
             (long)HEADS_ * N * N, (long)N * N, 0, 0,
             HEADS_, B_ * HEADS_, 1, 0, SCALE_);
        // softmaxes
        unsigned rows = (unsigned)(B_ * HEADS_ * N);
        softmax_rows<<<rows, 256, 0, stream>>>(d1, N);
        softmax_rows<<<rows, 256, 0, stream>>>(d2, N);
        softmax_rows<<<rows, 256, 0, stream>>>(d3, N);
        // t3 = a3·v + v     (merged-head output layout; residual = v slice)
        gemm(d3, qkv1 + 2 * INNER_, nullptr, qkv1 + 2 * INNER_, t3a, N, HD_, N,
             N, (int)Q4, INNER_, (int)Q4,
             (long)HEADS_ * N * N, (long)N * N, (long)N * Q4, HD_,
             (long)N * INNER_, HD_, (long)N * Q4, HD_,
             HEADS_, B_ * HEADS_, 0, 0, 1.f);
        gemm(d3, qkv2 + 2 * INNER_, nullptr, qkv2 + 2 * INNER_, t3b, N, HD_, N,
             N, (int)Q4, INNER_, (int)Q4,
             (long)HEADS_ * N * N, (long)N * N, (long)N * Q4, HD_,
             (long)N * INNER_, HD_, (long)N * Q4, HD_,
             HEADS_, B_ * HEADS_, 0, 0, 1.f);
        // o = a·t3  (merged heads)
        gemm(d1, t3a, nullptr, nullptr, o1m, N, HD_, N,
             N, INNER_, INNER_, 0,
             (long)HEADS_ * N * N, (long)N * N, (long)N * INNER_, HD_,
             (long)N * INNER_, HD_, 0, 0,
             HEADS_, B_ * HEADS_, 0, 0, 1.f);
        gemm(d2, t3b, nullptr, nullptr, o2m, N, HD_, N,
             N, INNER_, INNER_, 0,
             (long)HEADS_ * N * N, (long)N * N, (long)N * INNER_, HD_,
             (long)N * INNER_, HD_, 0, 0,
             HEADS_, B_ * HEADS_, 0, 0, 1.f);
        // out projection + residual (in place into X)
        gemm(o1m, in(pb + 4), in(pb + 5), X1, X1, (int)MN, DIM_, INNER_,
             INNER_, DIM_, DIM_, DIM_, 0, 0, 0, 0, 0, 0, 0, 0, 1, 1, 0, 0, 1.f);
        gemm(o2m, in(pb + 6), in(pb + 7), X2, X2, (int)MN, DIM_, INNER_,
             INNER_, DIM_, DIM_, DIM_, 0, 0, 0, 0, 0, 0, 0, 0, 1, 1, 0, 0, 1.f);
        // MLP stream 1
        layernorm_k<<<(unsigned)MN, 256, 0, stream>>>(X1, in(pb + 10), in(pb + 11), xn1, DIM_);
        gemm(xn1, in(pb + 12), in(pb + 13), nullptr, mlph, (int)MN, MLP_, DIM_,
             DIM_, MLP_, MLP_, 0, 0, 0, 0, 0, 0, 0, 0, 0, 1, 1, 0, 1, 1.f);
        gemm(mlph, in(pb + 14), in(pb + 15), X1, X1, (int)MN, DIM_, MLP_,
             MLP_, DIM_, DIM_, DIM_, 0, 0, 0, 0, 0, 0, 0, 0, 1, 1, 0, 0, 1.f);
        // MLP stream 2
        layernorm_k<<<(unsigned)MN, 256, 0, stream>>>(X2, in(pb + 10), in(pb + 11), xn2, DIM_);
        gemm(xn2, in(pb + 16), in(pb + 17), nullptr, mlph, (int)MN, MLP_, DIM_,
             DIM_, MLP_, MLP_, 0, 0, 0, 0, 0, 0, 0, 0, 0, 1, 1, 0, 1, 1.f);
        gemm(mlph, in(pb + 18), in(pb + 19), X2, X2, (int)MN, DIM_, MLP_,
             MLP_, DIM_, DIM_, DIM_, 0, 0, 0, 0, 0, 0, 0, 0, 1, 1, 0, 0, 1.f);
    };

    for (int l = 0; l < 4; ++l) runLayer(X11, X12, N1_,  48 + l * 20);
    for (int l = 0; l < 4; ++l) runLayer(X21, X22, N2_, 128 + l * 20);

    // ---- pooling + heads ----
    pool_k<<<blocks(B_ * DIM_), 256, 0, stream>>>(X11, X12, xspe, N1_);
    pool_k<<<blocks(B_ * DIM_), 256, 0, stream>>>(X21, X22, xspa, N2_);
    concat_k<<<blocks(B_ * 2 * DIM_), 256, 0, stream>>>(xspe, xspa, xcat);

    float* out = (float*)d_out;
    // head_spe
    gemm(xspe, in(208), in(209), nullptr, hh, B_, MLP_, DIM_,
         DIM_, MLP_, MLP_, 0, 0, 0, 0, 0, 0, 0, 0, 0, 1, 1, 0, 1, 1.f);
    gemm(hh, in(210), in(211), nullptr, out + 0, B_, 16, MLP_,
         MLP_, 16, 16, 0, 0, 0, 0, 0, 0, 0, 0, 0, 1, 1, 0, 0, 1.f);
    // head_spa
    gemm(xspa, in(212), in(213), nullptr, hh, B_, MLP_, DIM_,
         DIM_, MLP_, MLP_, 0, 0, 0, 0, 0, 0, 0, 0, 0, 1, 1, 0, 1, 1.f);
    gemm(hh, in(214), in(215), nullptr, out + 256, B_, 16, MLP_,
         MLP_, 16, 16, 0, 0, 0, 0, 0, 0, 0, 0, 0, 1, 1, 0, 0, 1.f);
    // head_t
    gemm(xcat, in(216), in(217), nullptr, hh, B_, MLP_, 2 * DIM_,
         2 * DIM_, MLP_, MLP_, 0, 0, 0, 0, 0, 0, 0, 0, 0, 1, 1, 0, 1, 1.f);
    gemm(hh, in(218), in(219), nullptr, out + 512, B_, 16, MLP_,
         MLP_, 16, 16, 0, 0, 0, 0, 0, 0, 0, 0, 0, 1, 1, 0, 0, 1.f);
}
